// G3N2Level_28750511080055
// MI455X (gfx1250) — matrix operand order repaired and checked by reading, not compiled, not run on hardware
//
#include <hip/hip_runtime.h>

// ---------------- problem constants (match reference) ----------------
constexpr int DIM  = 256;      // embed dim
constexpr int NN1  = 50000;    // nodes graph1 (3125 * 16)
constexpr int EE1  = 800000;   // edges graph1
constexpr int NDR  = 4096;     // drugs (256 * 16)
constexpr int EE2  = 131072;   // edges graph2
constexpr int L1N  = 3;
constexpr int L2N  = 2;

// ---------------- CDNA5 WMMA types ----------------
typedef __attribute__((ext_vector_type(16))) __bf16         v16bf;
typedef __attribute__((ext_vector_type(8)))  float          v8f;
typedef __attribute__((ext_vector_type(8)))  unsigned short v8us;

union BV {
  v8us           h[2];
  unsigned short us[16];
  v16bf          v;
};

union BCV {
  __bf16         b;
  unsigned short u;
};

// A operand: 16x32 bf16 tile, ISA 16-bit A layout.
// lane 0-15 : row=lane,    K = kb..kb+7 , kb+16..kb+23  (kb = s*32)
// lane16-31 : row=lane-16, K = kb+8..kb+15, kb+24..kb+31
__device__ __forceinline__ v16bf load_a(const float* __restrict__ rowp,
                                        float sc, int s, int lane) {
  const int kb = s * 32 + 8 * (lane >> 4);
  const float* p = rowp + kb;
  float4 f0 = *(const float4*)(p + 0);
  float4 f1 = *(const float4*)(p + 4);
  float4 f2 = *(const float4*)(p + 16);
  float4 f3 = *(const float4*)(p + 20);
  v16bf a;
  a[0]  = (__bf16)(f0.x * sc); a[1]  = (__bf16)(f0.y * sc);
  a[2]  = (__bf16)(f0.z * sc); a[3]  = (__bf16)(f0.w * sc);
  a[4]  = (__bf16)(f1.x * sc); a[5]  = (__bf16)(f1.y * sc);
  a[6]  = (__bf16)(f1.z * sc); a[7]  = (__bf16)(f1.w * sc);
  a[8]  = (__bf16)(f2.x * sc); a[9]  = (__bf16)(f2.y * sc);
  a[10] = (__bf16)(f2.z * sc); a[11] = (__bf16)(f2.w * sc);
  a[12] = (__bf16)(f3.x * sc); a[13] = (__bf16)(f3.y * sc);
  a[14] = (__bf16)(f3.z * sc); a[15] = (__bf16)(f3.w * sc);
  return a;
}

// B operand from pre-swizzled weights: contiguous 32B per lane, 1KB per tile.
__device__ __forceinline__ v16bf load_b(const unsigned short* __restrict__ Wpk,
                                        int nt, int s, int lane) {
  const unsigned short* p = Wpk + ((size_t)(((nt * 8 + s) * 32) + lane) << 4);
  BV b;
  b.h[0] = *(const v8us*)(p);
  b.h[1] = *(const v8us*)(p + 8);
  return b.v;
}

// A operand re-load from the LDS staging buffer (2 x ds_load_b128).
__device__ __forceinline__ v16bf lds_a(const unsigned short* smem,
                                       int tile, int lane) {
  const unsigned short* p = smem + ((size_t)tile * 32 + lane) * 16;
  BV b;
  b.h[0] = *(const v8us*)(p);
  b.h[1] = *(const v8us*)(p + 8);
  return b.v;
}

// out[M x 256] = relu( A*rs(cntA) @ WA + B*rs(cntB) @ WB [+ C @ WC] + bias )
// block = 128 threads (4 waves); block covers 16 rows x 256 cols; wave w owns
// N-tiles w*4..w*4+3. grid.x = M/16. NTERMS compile-time (2 or 3).
// Phase 1: cooperative f32->bf16 A-operand staging in LDS (each element
// converted exactly once per block instead of once per wave).
// Phase 2: fully unrolled K loop of WMMAs; A from LDS, B streamed from global.
template <int NTERMS>
__global__ __launch_bounds__(128)
void gemm_bf16_256(const float* __restrict__ A, const float* __restrict__ cntA,
                   const unsigned short* __restrict__ WA,
                   const float* __restrict__ B, const float* __restrict__ cntB,
                   const unsigned short* __restrict__ WB,
                   const float* __restrict__ C,
                   const unsigned short* __restrict__ WC,
                   const float* __restrict__ bias, float* __restrict__ out) {
  __shared__ unsigned short smem[NTERMS * 8 * 32 * 16];  // NTERMS*8 tiles, 1KB each

  const int mtile = blockIdx.x;
  const int wave  = threadIdx.x >> 5;
  const int lane  = threadIdx.x & 31;
  const int row   = mtile * 16 + (lane & 15);

  // per-row mean-normalization scales (fused), computed once
  const float scA = cntA ? (1.0f / fmaxf(cntA[row], 1.0f)) : 1.0f;
  const float scB = cntB ? (1.0f / fmaxf(cntB[row], 1.0f)) : 1.0f;

  // ---- phase 1: stage A operands into LDS, WMMA-ready layout ----
#pragma unroll
  for (int term = 0; term < NTERMS; ++term) {
    const float* base = (term == 0) ? A : (term == 1) ? B : C;
    const float  sc   = (term == 0) ? scA : (term == 1) ? scB : 1.0f;
    const float* rowp = base + (size_t)row * DIM;
#pragma unroll
    for (int si = 0; si < 2; ++si) {
      const int s = wave + 4 * si;               // each wave does 2 of 8 k-steps
      BV b;
      b.v = load_a(rowp, sc, s, lane);
      unsigned short* d = smem + ((size_t)(term * 8 + s) * 32 + lane) * 16;
      *(v8us*)(d)     = b.h[0];
      *(v8us*)(d + 8) = b.h[1];
    }
  }
  __syncthreads();

  // ---- phase 2: MAC loop ----
  v8f acc[4] = {};
#pragma unroll
  for (int s = 0; s < 8; ++s) {
    v16bf aA = lds_a(smem, 0 * 8 + s, lane);
    v16bf aB = lds_a(smem, 1 * 8 + s, lane);
    v16bf aC = aA;
    if (NTERMS == 3) aC = lds_a(smem, 2 * 8 + s, lane);
#pragma unroll
    for (int t = 0; t < 4; ++t) {
      const int nt = wave * 4 + t;
      acc[t] = __builtin_amdgcn_wmma_f32_16x16x32_bf16(
          false, aA, false, load_b(WA, nt, s, lane), (short)0, acc[t], false, false);
      acc[t] = __builtin_amdgcn_wmma_f32_16x16x32_bf16(
          false, aB, false, load_b(WB, nt, s, lane), (short)0, acc[t], false, false);
      if (NTERMS == 3)
        acc[t] = __builtin_amdgcn_wmma_f32_16x16x32_bf16(
            false, aC, false, load_b(WC, nt, s, lane), (short)0, acc[t], false, false);
    }
  }

  // C/D layout: lane 0-15 -> n=lane, rows m = v ; lane 16-31 -> rows m = 8+v
  const int rbase = mtile * 16 + 8 * (lane >> 4);
#pragma unroll
  for (int t = 0; t < 4; ++t) {
    const int ncol = (wave * 4 + t) * 16 + (lane & 15);
    const float bb = bias[ncol];
#pragma unroll
    for (int v = 0; v < 8; ++v) {
      float val = acc[t][v] + bb;
      out[(size_t)(rbase + v) * DIM + ncol] = fmaxf(val, 0.0f);
    }
  }
}

// Swizzle a 256x256 f32 weight (W[k*256+n]) into WMMA-B bf16 layout.
__global__ __launch_bounds__(256)
void pack_w_kernel(const float* __restrict__ W, unsigned short* __restrict__ Wpk) {
  const int idx  = blockIdx.x * 256 + threadIdx.x;  // 65536 total
  const int j    = idx & 15;
  const int lane = (idx >> 4) & 31;
  const int s    = (idx >> 9) & 7;
  const int t    = (idx >> 12);
  const int k = s * 32 + 8 * (lane >> 4) + (j & 7) + 16 * (j >> 3);
  const int n = t * 16 + (lane & 15);
  BCV cv;
  cv.b = (__bf16)W[k * DIM + n];
  Wpk[idx] = cv.u;
}

__global__ __launch_bounds__(256)
void zero_f(float* __restrict__ p, int n) {
  int i = blockIdx.x * 256 + threadIdx.x;
  if (i < n) p[i] = 0.0f;
}

__global__ __launch_bounds__(256)
void gather_rows(const int* __restrict__ ids, const float* __restrict__ tab,
                 float* __restrict__ out, int N) {
  int g = blockIdx.x * 256 + threadIdx.x;
  int i = g >> 6;
  if (i >= N) return;
  int c = (g & 63) << 2;
  int id = ids[i];
  *(float4*)(out + (size_t)i * DIM + c) = *(const float4*)(tab + (size_t)id * DIM + c);
}

__global__ __launch_bounds__(256)
void gather_drug(const int* __restrict__ ids, const float* __restrict__ xdrug,
                 const float* __restrict__ emb2, float* __restrict__ out, int N) {
  int g = blockIdx.x * 256 + threadIdx.x;
  int i = g >> 6;
  if (i >= N) return;
  int c = (g & 63) << 2;
  int id = ids[i];
  const float* srcp = (id < NDR) ? (xdrug + (size_t)id * DIM) : (emb2 + (size_t)id * DIM);
  *(float4*)(out + (size_t)i * DIM + c) = *(const float4*)(srcp + c);
}

__global__ __launch_bounds__(256)
void count_edges(const int* __restrict__ dst, float* __restrict__ cnt, int E) {
  int e = blockIdx.x * 256 + threadIdx.x;
  if (e < E) atomicAdd(cnt + dst[e], 1.0f);
}

// 64 threads per edge, float4 per thread: read x[src], atomic-add into agg[dst]
__global__ __launch_bounds__(256)
void scatter_edges(const int* __restrict__ src, const int* __restrict__ dst,
                   const float* __restrict__ x, float* __restrict__ agg, int E) {
  int g = blockIdx.x * 256 + threadIdx.x;
  int e = g >> 6;
  if (e >= E) return;
  int c = (g & 63) << 2;
  float4 v = *(const float4*)(x + (size_t)src[e] * DIM + c);
  float* p = agg + (size_t)dst[e] * DIM + c;
  atomicAdd(p + 0, v.x);
  atomicAdd(p + 1, v.y);
  atomicAdd(p + 2, v.z);
  atomicAdd(p + 3, v.w);
}

// segment max (uint-bit atomicMax; inputs are post-ReLU >= 0, init 0 also
// implements the isfinite->0 guard), segment sum, and node counts per drug.
__global__ __launch_bounds__(256)
void pool_kernel(const float* __restrict__ x, const int* __restrict__ batch,
                 float* __restrict__ gmax, float* __restrict__ gsum,
                 float* __restrict__ cntb, int N) {
  int g = blockIdx.x * 256 + threadIdx.x;
  int i = g >> 6;
  if (i >= N) return;
  int c = (g & 63) << 2;
  int b = batch[i];
  float4 v = *(const float4*)(x + (size_t)i * DIM + c);
  unsigned int* pm = (unsigned int*)(gmax + (size_t)b * DIM + c);
  atomicMax(pm + 0, __float_as_uint(v.x));
  atomicMax(pm + 1, __float_as_uint(v.y));
  atomicMax(pm + 2, __float_as_uint(v.z));
  atomicMax(pm + 3, __float_as_uint(v.w));
  float* ps = gsum + (size_t)b * DIM + c;
  atomicAdd(ps + 0, v.x);
  atomicAdd(ps + 1, v.y);
  atomicAdd(ps + 2, v.z);
  atomicAdd(ps + 3, v.w);
  if (c == 0) atomicAdd(cntb + b, 1.0f);
}

extern "C" void kernel_launch(void* const* d_in, const int* in_sizes, int n_in,
                              void* d_out, int out_size, void* d_ws, size_t ws_size,
                              hipStream_t stream) {
  (void)in_sizes; (void)n_in; (void)out_size; (void)ws_size;

  const int*   x1    = (const int*)d_in[0];
  const int*   ei1   = (const int*)d_in[1];   // [2, E1]: src = ei1, dst = ei1+E1
  const int*   batch = (const int*)d_in[2];
  const int*   x2ids = (const int*)d_in[3];
  const int*   ei2   = (const int*)d_in[4];   // [2, E2]
  const float* emb1  = (const float*)d_in[5];
  const float* emb2  = (const float*)d_in[6];
  const float* Wl1   = (const float*)d_in[7];
  const float* bl1   = (const float*)d_in[8];
  const float* Wr1   = (const float*)d_in[9];
  const float* lin1W = (const float*)d_in[10];  // [512,256]
  const float* lin1b = (const float*)d_in[11];
  const float* Wl2   = (const float*)d_in[12];
  const float* bl2   = (const float*)d_in[13];
  const float* Wr2   = (const float*)d_in[14];
  const float* Wx2   = (const float*)d_in[15];
  float* out = (float*)d_out;

  // ---------------- workspace layout ----------------
  float* ws = (float*)d_ws;
  size_t o = 0;
  float* xA   = ws + o; o += (size_t)NN1 * DIM;
  float* xB   = ws + o; o += (size_t)NN1 * DIM;
  float* agg1 = ws + o; o += (size_t)NN1 * DIM;
  float* cnt1 = ws + o; o += 51200;             // >= NN1, padded
  float* gmax = ws + o; o += (size_t)NDR * DIM;
  float* gsum = ws + o; o += (size_t)NDR * DIM;
  float* cntb = ws + o; o += 4096;
  float* xDrg = ws + o; o += (size_t)NDR * DIM;
  float* x2A  = ws + o; o += (size_t)NDR * DIM;
  float* x2B  = ws + o; o += (size_t)NDR * DIM;
  float* agg2 = ws + o; o += (size_t)NDR * DIM;
  float* cnt2 = ws + o; o += 4096;
  unsigned short* wpk = (unsigned short*)(ws + o);  // 14 packed 256x256 bf16 mats
  const size_t WSZ = (size_t)DIM * DIM;  // 65536 elems per matrix

  // ---- pack all weights to WMMA-B bf16 layout (tiny: 14 x 128KB) ----
  // slots: 0..2 Wl1, 3..5 Wr1, 6 lin1 top, 7 lin1 bottom,
  //        8..9 Wl2, 10..11 Wr2, 12..13 Wx2
  for (int i = 0; i < L1N; ++i)
    pack_w_kernel<<<256, 256, 0, stream>>>(Wl1 + i * WSZ, wpk + (0 + i) * WSZ);
  for (int i = 0; i < L1N; ++i)
    pack_w_kernel<<<256, 256, 0, stream>>>(Wr1 + i * WSZ, wpk + (3 + i) * WSZ);
  pack_w_kernel<<<256, 256, 0, stream>>>(lin1W,       wpk + 6 * WSZ);
  pack_w_kernel<<<256, 256, 0, stream>>>(lin1W + WSZ, wpk + 7 * WSZ);
  for (int i = 0; i < L2N; ++i)
    pack_w_kernel<<<256, 256, 0, stream>>>(Wl2 + i * WSZ, wpk + (8 + i) * WSZ);
  for (int i = 0; i < L2N; ++i)
    pack_w_kernel<<<256, 256, 0, stream>>>(Wr2 + i * WSZ, wpk + (10 + i) * WSZ);
  for (int i = 0; i < L2N; ++i)
    pack_w_kernel<<<256, 256, 0, stream>>>(Wx2 + i * WSZ, wpk + (12 + i) * WSZ);

  // ---- level 1 ----
  gather_rows<<<(NN1 * 64) / 256, 256, 0, stream>>>(x1, emb1, xA, NN1);

  zero_f<<<51200 / 256, 256, 0, stream>>>(cnt1, 51200);
  count_edges<<<(EE1 + 255) / 256, 256, 0, stream>>>(ei1 + EE1, cnt1, EE1);

  float* cur = xA;
  float* nxt = xB;
  for (int l = 0; l < L1N; ++l) {
    zero_f<<<((size_t)NN1 * DIM) / 256, 256, 0, stream>>>(agg1, NN1 * DIM);
    scatter_edges<<<((size_t)EE1 * 64) / 256, 256, 0, stream>>>(ei1, ei1 + EE1, cur, agg1, EE1);
    gemm_bf16_256<2><<<NN1 / 16, 128, 0, stream>>>(
        agg1, cnt1, wpk + (0 + l) * WSZ,
        cur, nullptr, wpk + (3 + l) * WSZ,
        nullptr, nullptr,
        bl1 + l * DIM, nxt);
    float* t = cur; cur = nxt; nxt = t;
  }

  // ---- pooling + lin1 ----
  zero_f<<<((size_t)NDR * DIM) / 256, 256, 0, stream>>>(gmax, NDR * DIM);
  zero_f<<<((size_t)NDR * DIM) / 256, 256, 0, stream>>>(gsum, NDR * DIM);
  zero_f<<<4096 / 256, 256, 0, stream>>>(cntb, 4096);
  pool_kernel<<<(NN1 * 64) / 256, 256, 0, stream>>>(cur, batch, gmax, gsum, cntb, NN1);
  // xDrug = relu(gmax @ Wtop + (gsum/max(cnt,1)) @ Wbot + b)
  gemm_bf16_256<2><<<NDR / 16, 128, 0, stream>>>(
      gmax, nullptr, wpk + 6 * WSZ,
      gsum, cntb,    wpk + 7 * WSZ,
      nullptr, nullptr,
      lin1b, xDrg);

  // ---- level 2 ----
  gather_drug<<<(NDR * 64) / 256, 256, 0, stream>>>(x2ids, xDrg, emb2, x2A, NDR);
  zero_f<<<4096 / 256, 256, 0, stream>>>(cnt2, 4096);
  count_edges<<<(EE2 + 255) / 256, 256, 0, stream>>>(ei2 + EE2, cnt2, EE2);

  float* c2 = x2A;
  for (int l = 0; l < L2N; ++l) {
    zero_f<<<((size_t)NDR * DIM) / 256, 256, 0, stream>>>(agg2, NDR * DIM);
    scatter_edges<<<((size_t)EE2 * 64) / 256, 256, 0, stream>>>(ei2, ei2 + EE2, c2, agg2, EE2);
    float* dst = (l == L2N - 1) ? out : x2B;
    gemm_bf16_256<3><<<NDR / 16, 128, 0, stream>>>(
        agg2, cnt2,  wpk + (8 + l) * WSZ,
        c2, nullptr, wpk + (10 + l) * WSZ,
        xDrg, wpk + (12 + l) * WSZ,
        bl2 + l * DIM, dst);
    c2 = dst;
  }
}
